// GATExplainer_62242666054091
// MI455X (gfx1250) — matrix operand order repaired and checked by reading, compiled, not run on hardware
//
#include <hip/hip_runtime.h>
#include <hip/hip_bf16.h>

// ---------------------------------------------------------------------------
// GATExplainer pipeline for MI455X (gfx1250, wave32).
// GEMMs: v_wmma_f32_16x16x32_bf16. B is staged in LDS (bf16, column-major,
// bank-padded) once per block -> B fragments are 2x ds_load_b128 per WMMA.
// A fragments come from aligned global_load_b128/b64. One wave owns a 16x64
// output strip (4 col-tiles) so the A fragment feeds 4 WMMAs per K-step.
// Edge phase: L2-resident gathers + global f32/u32 atomics (192MB L2 holds
// all node feature tables), 3-pass segment softmax.
// ---------------------------------------------------------------------------

typedef __bf16 v16bf __attribute__((ext_vector_type(16)));
typedef __bf16 v8bf  __attribute__((ext_vector_type(8)));
typedef float  v8f   __attribute__((ext_vector_type(8)));

#define NEG_SLOPE 0.2f
#define BN_EPS 1e-5f

// --------------------------- WMMA GEMM -------------------------------------
// C[M x NC] = A[M x K] * B[K x NC], f32 in memory, bf16 MACs.
// Requires M % 16 == 0, NC % 64 == 0, blockDim.x == 256.
// Fragment layouts per CDNA5 ISA 7.12.2 (wave32):
//   A (16x32 bf16): lane&15 = row; lanes<16 hold K=kb..kb+7 with kb=0 and
//                   16..23; lanes>=16 hold 8..15 and 24..31.
//   B (32x16 bf16): lane&15 = col; lanes<16 hold K=0..7 and 16..23, lanes>=16 8..15 and 24..31 (as A).
//   C/D (16x16 f32): VGPR r: lanes<16 -> M=r, lanes>=16 -> M=8+r, N=lane&15.
template <int K, int NC>
__global__ void gat_gemm_wmma(const float* __restrict__ A,
                              const float* __restrict__ B,
                              float* __restrict__ C, int M) {
  constexpr int KP   = (K + 31) & ~31;   // K padded to 32
  constexpr int LDSK = KP + 8;           // +16B pad -> conflict-spread columns
  __shared__ __bf16 Bs[NC * LDSK];       // B^T, bf16: Bs[n*LDSK + k]

  // ---- stage all of B into LDS (all 256 threads; tiny: <=34KB) ----
  for (int idx = threadIdx.x; idx < NC * KP; idx += 256) {
    int n = idx / KP;
    int k = idx & (KP - 1);
    float v = B[(size_t)(k < K ? k : K - 1) * NC + n];
    Bs[n * LDSK + k] = (__bf16)(k < K ? v : 0.0f);
  }
  __syncthreads();

  int wave = (blockIdx.x * blockDim.x + threadIdx.x) >> 5;
  int lane = threadIdx.x & 31;
  constexpr int groupsN = NC >> 6;       // one wave = 4 col-tiles = 64 cols
  int tilesM = M >> 4;                   // M % 16 == 0 guaranteed by host
  if (wave >= tilesM * groupsN) return;
  int tm = wave / groupsN;
  int g  = wave - tm * groupsN;

  int row  = tm * 16 + (lane & 15);
  int col0 = g * 64 + (lane & 15);
  int kbA  = (lane >> 4) * 8;
  int kbB  = (lane >> 4) * 8;

  const float* __restrict__ Arow = A + (size_t)row * K;

  v8f acc0 = {}, acc1 = {}, acc2 = {}, acc3 = {};

  for (int k0 = 0; k0 < K; k0 += 32) {
    // ---- A fragment (shared by the 4 WMMAs of this K-step) ----
    v16bf a;
    if ((K & 31) == 0) {
      // fully in-bounds, 16B-aligned: 4x global_load_b128
      const float4* ap = reinterpret_cast<const float4*>(Arow + k0 + kbA);
      float4 x0 = ap[0];
      float4 x1 = ap[1];
      const float4* aq = reinterpret_cast<const float4*>(Arow + k0 + 16 + kbA);
      float4 x2 = aq[0];
      float4 x3 = aq[1];
      a[0]  = (__bf16)x0.x; a[1]  = (__bf16)x0.y; a[2]  = (__bf16)x0.z; a[3]  = (__bf16)x0.w;
      a[4]  = (__bf16)x1.x; a[5]  = (__bf16)x1.y; a[6]  = (__bf16)x1.z; a[7]  = (__bf16)x1.w;
      a[8]  = (__bf16)x2.x; a[9]  = (__bf16)x2.y; a[10] = (__bf16)x2.z; a[11] = (__bf16)x2.w;
      a[12] = (__bf16)x3.x; a[13] = (__bf16)x3.y; a[14] = (__bf16)x3.z; a[15] = (__bf16)x3.w;
    } else {
      // K < 16 path (K even): elements 0..7 from 4 aligned float2 loads with
      // clamped addresses; elements 8..15 have k = 16+kbA+i >= 16 > K -> 0.
#pragma unroll
      for (int i = 0; i < 8; i += 2) {
        int k = kbA + i;
        const float2 pr =
            *reinterpret_cast<const float2*>(Arow + (k + 1 < K ? k : K - 2));
        a[i]     = (__bf16)(k     < K ? pr.x : 0.0f);
        a[i + 1] = (__bf16)(k + 1 < K ? pr.y : 0.0f);
      }
#pragma unroll
      for (int i = 8; i < 16; ++i) a[i] = (__bf16)0.0f;
    }
    // ---- 4 B fragments from LDS (2x ds_load_b128 each), 4 WMMAs ----
#pragma unroll
    for (int j = 0; j < 4; ++j) {
      const __bf16* bp = &Bs[(size_t)(col0 + j * 16) * LDSK + k0 + kbB];
      v8bf blo = *reinterpret_cast<const v8bf*>(bp);      // K = kbB+0..7
      v8bf bhi = *reinterpret_cast<const v8bf*>(bp + 16); // K = kbB+16..23
      v16bf b = __builtin_shufflevector(blo, bhi,
                                        0, 1, 2, 3, 4, 5, 6, 7,
                                        8, 9, 10, 11, 12, 13, 14, 15);
      v8f& acc = (j == 0) ? acc0 : (j == 1) ? acc1 : (j == 2) ? acc2 : acc3;
      acc = __builtin_amdgcn_wmma_f32_16x16x32_bf16(
          /*neg_a=*/false, a, /*neg_b=*/false, b,
          /*c_mod=*/(short)0, acc, /*reuse_a=*/false, /*reuse_b=*/false);
    }
  }

  int rbase = tm * 16 + ((lane >> 4) * 8);
#pragma unroll
  for (int j = 0; j < 4; ++j) {
    const v8f& acc = (j == 0) ? acc0 : (j == 1) ? acc1 : (j == 2) ? acc2 : acc3;
#pragma unroll
    for (int r = 0; r < 8; ++r)
      C[(size_t)(rbase + r) * NC + col0 + j * 16] = acc[r];
  }
}

// ------------------- attention alpha coefficients --------------------------
// as[n,h] = sum_c h[n,h,c]*a_src[h,c] ; ad likewise.
__global__ void gat_alpha(const float* __restrict__ h,
                          const float* __restrict__ a_src,
                          const float* __restrict__ a_dst,
                          float* __restrict__ as, float* __restrict__ ad,
                          int N, int H, int C) {
  int t = blockIdx.x * blockDim.x + threadIdx.x;
  if (t >= N * H) return;
  int n = t / H, hh = t - n * H;
  const float* rowp = h + (size_t)n * H * C + (size_t)hh * C;
  const float* s = a_src + (size_t)hh * C;
  const float* d = a_dst + (size_t)hh * C;
  float s1 = 0.f, s2 = 0.f;
  for (int c = 0; c < C; ++c) { float v = rowp[c]; s1 += v * s[c]; s2 += v * d[c]; }
  as[t] = s1; ad[t] = s2;
}

// ------------------- order-preserving float<->u32 for atomicMax ------------
__device__ __forceinline__ unsigned fenc(float f) {
  unsigned u = __float_as_uint(f);
  return (u & 0x80000000u) ? ~u : (u | 0x80000000u);
}
__device__ __forceinline__ float fdec(unsigned e) {
  unsigned u = (e & 0x80000000u) ? (e ^ 0x80000000u) : ~e;
  return __uint_as_float(u);
}

__device__ __forceinline__ void edge_sd(const int* __restrict__ ei, int E, int e,
                                        int& s, int& d) {
  if (e < E) { s = ei[e]; d = ei[E + e]; }     // edge_index rows [src; dst]
  else       { s = d = e - E; }                // appended self-loop
}

// pass 1: segment max of leaky_relu(as[src]+ad[dst]) over dst
__global__ void gat_edge_max(const int* __restrict__ ei, int E, int N, int H,
                             const float* __restrict__ as,
                             const float* __restrict__ ad,
                             unsigned* __restrict__ mx) {
  int t = blockIdx.x * blockDim.x + threadIdx.x;
  int tot = (E + N) * H;
  if (t >= tot) return;
  int h = t % H, e = t / H, s, d;
  edge_sd(ei, E, e, s, d);
  float v = as[s * H + h] + ad[d * H + h];
  v = v > 0.f ? v : NEG_SLOPE * v;
  atomicMax(&mx[d * H + h], fenc(v));
}

// pass 2: denom[dst,h] += exp(e - m[dst,h])
__global__ void gat_edge_sum(const int* __restrict__ ei, int E, int N, int H,
                             const float* __restrict__ as,
                             const float* __restrict__ ad,
                             const unsigned* __restrict__ mx,
                             float* __restrict__ den) {
  int t = blockIdx.x * blockDim.x + threadIdx.x;
  int tot = (E + N) * H;
  if (t >= tot) return;
  int h = t % H, e = t / H, s, d;
  edge_sd(ei, E, e, s, d);
  float v = as[s * H + h] + ad[d * H + h];
  v = v > 0.f ? v : NEG_SLOPE * v;
  float m = fdec(mx[d * H + h]);
  atomicAdd(&den[d * H + h], __expf(v - m));
}

// pass 3: out[dst, h, c] += h[src, h, c] * alpha(e,h)   (4 channels/thread)
__global__ void gat_edge_msg(const int* __restrict__ ei, int E, int N, int H, int C,
                             const float* __restrict__ hbuf,
                             const float* __restrict__ as,
                             const float* __restrict__ ad,
                             const unsigned* __restrict__ mx,
                             const float* __restrict__ den,
                             float* __restrict__ out) {
  int CH = H * C;
  int per = CH >> 2;                       // 4 channels per thread
  int t = blockIdx.x * blockDim.x + threadIdx.x;
  if (t >= (E + N) * per) return;
  int e = t / per;
  int ch = (t - e * per) << 2;
  int h = ch / C;
  int s, d;
  edge_sd(ei, E, e, s, d);
  float v = as[s * H + h] + ad[d * H + h];
  v = v > 0.f ? v : NEG_SLOPE * v;
  float m = fdec(mx[d * H + h]);
  float alpha = __expf(v - m) / (den[d * H + h] + 1e-16f);
  const float4 hv = *reinterpret_cast<const float4*>(hbuf + (size_t)s * CH + ch);
  float* op = out + (size_t)d * CH + ch;
  atomicAdd(op + 0, hv.x * alpha);
  atomicAdd(op + 1, hv.y * alpha);
  atomicAdd(op + 2, hv.z * alpha);
  atomicAdd(op + 3, hv.w * alpha);
}

// ------------------------------- BatchNorm ---------------------------------
// blockDim.x == F; each block reduces a row-slice, then one atomic per column.
__global__ void gat_bn_reduce(const float* __restrict__ in,
                              const float* __restrict__ bias,
                              int N, int F, float* __restrict__ acc) {
  int f = threadIdx.x;
  int per = (N + gridDim.x - 1) / gridDim.x;
  int r0 = blockIdx.x * per;
  int r1 = r0 + per; if (r1 > N) r1 = N;
  float b = bias[f];
  float s = 0.f, s2 = 0.f;
  for (int r = r0; r < r1; ++r) {
    float v = in[(size_t)r * F + f] + b;
    s += v; s2 += v * v;
  }
  atomicAdd(&acc[f], s);
  atomicAdd(&acc[F + f], s2);
}

// y = elu(gamma*(x+b - mu)*rsqrt(var+eps) + beta), in place.
__global__ void gat_bn_apply_elu(float* __restrict__ x,
                                 const float* __restrict__ bias,
                                 const float* __restrict__ gamma,
                                 const float* __restrict__ beta,
                                 const float* __restrict__ acc,
                                 int N, int F) {
  int t = blockIdx.x * blockDim.x + threadIdx.x;
  if (t >= N * F) return;
  int f = t % F;
  float invN = 1.0f / (float)N;
  float mu = acc[f] * invN;
  float var = acc[F + f] * invN - mu * mu;
  float y = (x[t] + bias[f] - mu) * rsqrtf(var + BN_EPS) * gamma[f] + beta[f];
  x[t] = y > 0.f ? y : (__expf(y) - 1.0f);
}

// ------------------------------ mean pool ----------------------------------
__global__ void gat_pool(const float* __restrict__ x, const int* __restrict__ batch,
                         int N, int F, float* __restrict__ sums,
                         float* __restrict__ counts) {
  int t = blockIdx.x * blockDim.x + threadIdx.x;
  if (t >= N * F) return;
  int n = t / F, f = t - n * F;
  int g = batch[n];
  atomicAdd(&sums[(size_t)g * F + f], x[t]);
  if (f == 0) atomicAdd(&counts[g], 1.0f);
}

// ------------------------------ classifier ---------------------------------
// One wave per graph: lane j owns hidden unit j (32 lanes == 32 hidden units).
__global__ void gat_classifier(const float* __restrict__ sums,
                               const float* __restrict__ counts,
                               const float* __restrict__ Wc1,
                               const float* __restrict__ bc1,
                               const float* __restrict__ Wc2,
                               const float* __restrict__ bc2,
                               float* __restrict__ out, int B) {
  int wave = (blockIdx.x * blockDim.x + threadIdx.x) >> 5;
  int lane = threadIdx.x & 31;
  if (wave >= B) return;
  float inv = 1.0f / fmaxf(counts[wave], 1.0f);
  float hid = bc1[lane];
  const float* gv = sums + (size_t)wave * 64;
  for (int k = 0; k < 64; ++k) hid += gv[k] * inv * Wc1[k * 32 + lane];
  hid = fmaxf(hid, 0.0f);
#pragma unroll
  for (int o = 0; o < 2; ++o) {
    float p = hid * Wc2[lane * 2 + o];
    for (int off = 16; off >= 1; off >>= 1) p += __shfl_xor(p, off, 32);
    if (lane == 0) out[wave * 2 + o] = p + bc2[o];
  }
}

// ------------------------------- launch ------------------------------------
static inline unsigned nblk(long long n, int b) { return (unsigned)((n + b - 1) / b); }

extern "C" void kernel_launch(void* const* d_in, const int* in_sizes, int n_in,
                              void* d_out, int out_size, void* d_ws, size_t ws_size,
                              hipStream_t stream) {
  const float* x    = (const float*)d_in[0];
  const int*   ei   = (const int*)d_in[1];
  const int*   batch= (const int*)d_in[2];
  const float* W1   = (const float*)d_in[3];
  const float* a1s  = (const float*)d_in[4];
  const float* a1d  = (const float*)d_in[5];
  const float* b1   = (const float*)d_in[6];
  const float* g1   = (const float*)d_in[7];
  const float* be1  = (const float*)d_in[8];
  const float* W2   = (const float*)d_in[9];
  const float* a2s  = (const float*)d_in[10];
  const float* a2d  = (const float*)d_in[11];
  const float* b2   = (const float*)d_in[12];
  const float* g2   = (const float*)d_in[13];
  const float* be2  = (const float*)d_in[14];
  const float* Wc1  = (const float*)d_in[15];
  const float* bc1  = (const float*)d_in[16];
  const float* Wc2  = (const float*)d_in[17];
  const float* bc2  = (const float*)d_in[18];
  float* out = (float*)d_out;

  const int N = in_sizes[0] / 10;       // 50000  (multiple of 16)
  const int E = in_sizes[1] / 2;        // 800000
  const int B = 2500;
  const int Etot = E + N;               // + self loops

  // ---- workspace layout (f32 elements) ----
  float* p = (float*)d_ws;
  float*    h1   = p;            p += (size_t)N * 256;   // layer1 pre-agg features
  float*    h2   = p;            p += (size_t)N * 64;    // layer2 pre-agg features
  float*    as1  = p;            p += (size_t)N * 4;
  float*    ad1  = p;            p += (size_t)N * 4;
  float*    as2  = p;            p += N;
  float*    ad2  = p;            p += N;
  float*    zs   = p;                                    // ---- zeroed block ----
  float*    out1 = p;            p += (size_t)N * 256;
  float*    out2 = p;            p += (size_t)N * 64;
  unsigned* mx1  = (unsigned*)p; p += (size_t)N * 4;
  float*    den1 = p;            p += (size_t)N * 4;
  unsigned* mx2  = (unsigned*)p; p += N;
  float*    den2 = p;            p += N;
  float*    acc1 = p;            p += 512;               // 256 sums + 256 sumsq
  float*    acc2 = p;            p += 128;               // 64 + 64
  float*    pool = p;            p += (size_t)B * 64;
  float*    cnts = p;            p += B;
  size_t zero_bytes = (size_t)(p - zs) * sizeof(float);

  hipMemsetAsync(zs, 0, zero_bytes, stream);             // capture-safe

  const int T = 256;

  // ================= Layer 1 (H=4, C=64) =================
  // h1 = x @ W1 : M=N, K=10 (padded to 32), NC=256 -> WMMA bf16
  {
    long long waves = (long long)(N / 16) * (256 / 64);
    gat_gemm_wmma<10, 256><<<nblk(waves * 32, T), T, 0, stream>>>(x, W1, h1, N);
  }
  gat_alpha<<<nblk((long long)N * 4, T), T, 0, stream>>>(h1, a1s, a1d, as1, ad1, N, 4, 64);
  gat_edge_max<<<nblk((long long)Etot * 4, T), T, 0, stream>>>(ei, E, N, 4, as1, ad1, mx1);
  gat_edge_sum<<<nblk((long long)Etot * 4, T), T, 0, stream>>>(ei, E, N, 4, as1, ad1, mx1, den1);
  gat_edge_msg<<<nblk((long long)Etot * 64, T), T, 0, stream>>>(ei, E, N, 4, 64, h1, as1, ad1, mx1, den1, out1);
  gat_bn_reduce<<<256, 256, 0, stream>>>(out1, b1, N, 256, acc1);
  gat_bn_apply_elu<<<nblk((long long)N * 256, T), T, 0, stream>>>(out1, b1, g1, be1, acc1, N, 256);

  // ================= Layer 2 (H=1, C=64) =================
  // h2 = x1 @ W2 : M=N, K=256, NC=64 -> WMMA bf16 (8 K-steps, b128 A loads)
  {
    long long waves = (long long)(N / 16) * (64 / 64);
    gat_gemm_wmma<256, 64><<<nblk(waves * 32, T), T, 0, stream>>>(out1, W2, h2, N);
  }
  gat_alpha<<<nblk((long long)N, T), T, 0, stream>>>(h2, a2s, a2d, as2, ad2, N, 1, 64);
  gat_edge_max<<<nblk((long long)Etot, T), T, 0, stream>>>(ei, E, N, 1, as2, ad2, mx2);
  gat_edge_sum<<<nblk((long long)Etot, T), T, 0, stream>>>(ei, E, N, 1, as2, ad2, mx2, den2);
  gat_edge_msg<<<nblk((long long)Etot * 16, T), T, 0, stream>>>(ei, E, N, 1, 64, h2, as2, ad2, mx2, den2, out2);
  gat_bn_reduce<<<256, 64, 0, stream>>>(out2, b2, N, 64, acc2);
  gat_bn_apply_elu<<<nblk((long long)N * 64, T), T, 0, stream>>>(out2, b2, g2, be2, acc2, N, 64);

  // ================= pool + classifier =================
  gat_pool<<<nblk((long long)N * 64, T), T, 0, stream>>>(out2, batch, N, 64, pool, cnts);
  gat_classifier<<<nblk((long long)B * 32, T), T, 0, stream>>>(pool, cnts, Wc1, bc1, Wc2, bc2, out, B);
}
